// Net_90744069030471
// MI455X (gfx1250) — matrix operand order repaired and checked by reading, not compiled, not run on hardware
//
#include <hip/hip_runtime.h>
#include <math.h>

// ---------------- problem constants (from reference) ----------------
#define N_NODES 30000
#define E_EDGES 960000
#define NHEAD   2
#define NCHAN   128
#define NGRAPH  64
#define DIN     448
#define DHID    256   // NHEAD*NCHAN
#define DMLP    128
#define BN_EPS  1e-5f
#define BM      80    // M rows per block (5 WMMA tiles); 30000 = 375*80

// ---------------- CDNA5 WMMA types ----------------
typedef __attribute__((ext_vector_type(16))) __bf16 v16bf;
typedef __attribute__((ext_vector_type(8)))  float  v8f;

// ---------------- gfx1250 async copy (memory -> LDS), ASYNCcnt-tracked ----
__device__ __forceinline__ void async_copy_b128(unsigned lds_off, const float* gaddr) {
    asm volatile("global_load_async_to_lds_b128 %0, %1, off"
                 :: "v"(lds_off), "v"(gaddr)
                 : "memory");
}
__device__ __forceinline__ void wait_async0() {
    asm volatile("s_wait_asynccnt 0" ::: "memory");
}

// stage a 32 x 128 fp32 slice of a row-major weight matrix into LDS:
//   bsm[row][col] = src[(kb+row)*ldsrc + ncol0 + col]
// 256 threads x 4 float4 copies each = 1024 * 16B = 16 KB
__device__ __forceinline__ void stage_b_slice(const float* __restrict__ src,
                                              int ldsrc, int kb, int ncol0,
                                              float (*bsm)[128], int tid) {
#pragma unroll
    for (int j = 0; j < 4; ++j) {
        int f   = tid + j * 256;      // float4 id 0..1023
        int row = f >> 5;             // 32 float4 per 128-col row
        int c4  = (f & 31) * 4;
        unsigned lds = (unsigned)(size_t)&bsm[row][c4];   // low 32b of generic = LDS offset
        const float* g = src + (size_t)(kb + row) * ldsrc + ncol0 + c4;
        async_copy_b128(lds, g);
    }
}

// order-preserving float<->uint key (for atomic max on floats)
__device__ __forceinline__ unsigned f2key(float f) {
    unsigned b = __float_as_uint(f);
    return (b & 0x80000000u) ? ~b : (b | 0x80000000u);
}
__device__ __forceinline__ float key2f(unsigned u) {
    unsigned b = (u & 0x80000000u) ? (u & 0x7fffffffu) : ~u;
    return __uint_as_float(b);
}

// =====================================================================
// K0: initialize accumulators in workspace (must be re-done every call)
// =====================================================================
__global__ void init_ws_kernel(unsigned* __restrict__ mkey,
                               float* __restrict__ denom,
                               float* __restrict__ outg,
                               float* __restrict__ hp) {
    int i = blockIdx.x * blockDim.x + threadIdx.x;
    if (i < N_NODES * NHEAD) {
        mkey[i]  = 0x007FFFFFu;   // f2key(-inf)
        denom[i] = 0.0f;
    }
    if (i < N_NODES * DHID) outg[i] = 0.0f;
    if (i < NGRAPH * DMLP)  hp[i]   = 0.0f;
}

// =====================================================================
// K1: xp = x @ gat_w   [30000,448]@[448,256]  bf16 WMMA, fp32 accum
// block: 256 thr (8 waves); block tile 80(M) x 128(N); grid (375, 2)
// B K-slice double-buffered in LDS via global_load_async_to_lds_b128
// =====================================================================
__global__ __launch_bounds__(256)
void gemm_xp_kernel(const float* __restrict__ x,
                    const float* __restrict__ w,
                    float* __restrict__ xp) {
    __shared__ float bsm[2][32][128];
    const int tid  = threadIdx.x;
    const int lane = tid & 31;
    const int wv   = tid >> 5;            // wave = N-tile within block (0..7)
    const int kgrp = lane >> 4;
    const int m0   = blockIdx.x * BM;
    const int nc0  = blockIdx.y * 128;
    const int col  = nc0 + wv * 16 + (lane & 15);
    const int arow = lane & 15;

    stage_b_slice(w, DHID, 0, nc0, bsm[0], tid);

    v8f acc[5] = {{}, {}, {}, {}, {}};
    const int KS = DIN / 32;              // 14
    for (int s = 0; s < KS; ++s) {
        const int buf = s & 1, kb = s * 32;
        wait_async0();                    // own async copies landed
        __syncthreads();                  // everyone's copies landed
        if (s + 1 < KS) stage_b_slice(w, DHID, kb + 32, nc0, bsm[buf ^ 1], tid);

        // B fragment from LDS (col = lane&15, K = 8*kgrp + i for i<8, 16 + 8*kgrp + (i-8) for i>=8)
        v16bf bf;
#pragma unroll
        for (int i = 0; i < 16; ++i)
            bf[i] = (__bf16)bsm[buf][kgrp * 8 + ((i < 8) ? i : (8 + i))][wv * 16 + (lane & 15)];

        // 5 M-tiles reuse one B fragment -> 5 WMMAs per K-step
#pragma unroll
        for (int mt = 0; mt < 5; ++mt) {
            const float* ap = x + (size_t)(m0 + mt * 16 + arow) * DIN + kb + kgrp * 8;
            float af[16];
            *(float4*)(af + 0)  = *(const float4*)(ap);
            *(float4*)(af + 4)  = *(const float4*)(ap + 4);
            *(float4*)(af + 8)  = *(const float4*)(ap + 16);
            *(float4*)(af + 12) = *(const float4*)(ap + 20);
            v16bf a;
#pragma unroll
            for (int i = 0; i < 16; ++i) a[i] = (__bf16)af[i];
            acc[mt] = __builtin_amdgcn_wmma_f32_16x16x32_bf16(
                false, a, false, bf, (short)0, acc[mt], false, false);
        }
    }
#pragma unroll
    for (int mt = 0; mt < 5; ++mt)
#pragma unroll
        for (int r = 0; r < 8; ++r)
            xp[(size_t)(m0 + mt * 16 + r + kgrp * 8) * DHID + col] = acc[mt][r];
}

// =====================================================================
// K2: per-node attention logits  a_s[n,h], a_d[n,h]
// =====================================================================
__global__ void attn_dots_kernel(const float* __restrict__ xp,
                                 const float* __restrict__ asrc,
                                 const float* __restrict__ adst,
                                 float* __restrict__ a_s,
                                 float* __restrict__ a_d) {
    int idx = blockIdx.x * blockDim.x + threadIdx.x;   // node*2 + head
    if (idx >= N_NODES * NHEAD) return;
    int n = idx >> 1, h = idx & 1;
    const float* xr = xp + (size_t)n * DHID + h * NCHAN;
    const float* as = asrc + h * NCHAN;
    const float* ad = adst + h * NCHAN;
    float s1 = 0.f, s2 = 0.f;
    for (int c = 0; c < NCHAN; ++c) {
        float v = xr[c];
        s1 += v * as[c];
        s2 += v * ad[c];
    }
    a_s[idx] = s1;
    a_d[idx] = s2;
}

// =====================================================================
// K3/K4: edge softmax passes (self-loops are edges E..E+N-1)
// =====================================================================
__device__ __forceinline__ void edge_endpoints(int e, const int* __restrict__ ei,
                                               int& src, int& dst) {
    if (e < E_EDGES) { src = ei[e]; dst = ei[E_EDGES + e]; }
    else             { src = dst = e - E_EDGES; }
}

__global__ void edge_max_kernel(const int* __restrict__ ei,
                                const float* __restrict__ a_s,
                                const float* __restrict__ a_d,
                                unsigned* __restrict__ mkey) {
    int idx = blockIdx.x * blockDim.x + threadIdx.x;   // edge*2 + head
    if (idx >= (E_EDGES + N_NODES) * NHEAD) return;
    int e = idx >> 1, h = idx & 1;
    int src, dst; edge_endpoints(e, ei, src, dst);
    float s = a_s[src * NHEAD + h] + a_d[dst * NHEAD + h];
    s = (s >= 0.f) ? s : 0.2f * s;                     // leaky_relu(0.2)
    atomicMax(&mkey[dst * NHEAD + h], f2key(s));
}

__global__ void edge_exp_kernel(const int* __restrict__ ei,
                                const float* __restrict__ a_s,
                                const float* __restrict__ a_d,
                                const unsigned* __restrict__ mkey,
                                float* __restrict__ ew,
                                float* __restrict__ denom) {
    int idx = blockIdx.x * blockDim.x + threadIdx.x;
    if (idx >= (E_EDGES + N_NODES) * NHEAD) return;
    int e = idx >> 1, h = idx & 1;
    int src, dst; edge_endpoints(e, ei, src, dst);
    float s = a_s[src * NHEAD + h] + a_d[dst * NHEAD + h];
    s = (s >= 0.f) ? s : 0.2f * s;
    float m = key2f(mkey[dst * NHEAD + h]);
    float wv = __expf(s - m);
    ew[idx] = wv;
    atomicAdd(&denom[dst * NHEAD + h], wv);
}

// =====================================================================
// K5: aggregation — one wave per edge; gather xp[src] row (L2-resident),
//     scale per-head, atomic-add into out[dst]
// =====================================================================
__global__ __launch_bounds__(256)
void aggregate_kernel(const int* __restrict__ ei,
                      const float* __restrict__ xp,
                      const float* __restrict__ ew,
                      const float* __restrict__ denom,
                      float* __restrict__ outg) {
    int wid  = (blockIdx.x * blockDim.x + threadIdx.x) >> 5;   // edge id
    int lane = threadIdx.x & 31;
    if (wid >= E_EDGES + N_NODES) return;
    int src, dst; edge_endpoints(wid, ei, src, dst);

    float al0 = ew[wid * 2 + 0] / denom[dst * 2 + 0];
    float al1 = ew[wid * 2 + 1] / denom[dst * 2 + 1];

    const float4* xr = (const float4*)(xp + (size_t)src * DHID);
    float* orow = outg + (size_t)dst * DHID;
    __builtin_prefetch(xp + (size_t)src * DHID, 0, 1);   // global_prefetch_b8

#pragma unroll
    for (int j = 0; j < 2; ++j) {
        int c4 = lane + j * 32;             // float4 index 0..63 (64*4 = 256 ch)
        float4 v = xr[c4];
        float al = (c4 < 32) ? al0 : al1;   // head split at channel 128
        atomicAdd(orow + c4 * 4 + 0, al * v.x);
        atomicAdd(orow + c4 * 4 + 1, al * v.y);
        atomicAdd(orow + c4 * 4 + 2, al * v.z);
        atomicAdd(orow + c4 * 4 + 3, al * v.w);
    }
}

// =====================================================================
// K6: h = elu(out + gat_bias) @ mlp_w + mlp_b   (bf16 WMMA, fused ELU)
// block tile 80(M) x 128(N=all); grid (375, 1); B staged async in LDS
// =====================================================================
__global__ __launch_bounds__(256)
void gemm_h_kernel(const float* __restrict__ outg,
                   const float* __restrict__ gbias,
                   const float* __restrict__ w,
                   const float* __restrict__ mb,
                   float* __restrict__ hbuf) {
    __shared__ float bsm[2][32][128];
    const int tid  = threadIdx.x;
    const int lane = tid & 31;
    const int wv   = tid >> 5;
    const int kgrp = lane >> 4;
    const int m0   = blockIdx.x * BM;
    const int col  = wv * 16 + (lane & 15);
    const int arow = lane & 15;

    stage_b_slice(w, DMLP, 0, 0, bsm[0], tid);

    v8f acc[5] = {{}, {}, {}, {}, {}};
    const int KS = DHID / 32;             // 8
    for (int s = 0; s < KS; ++s) {
        const int buf = s & 1, kb = s * 32;
        wait_async0();
        __syncthreads();
        if (s + 1 < KS) stage_b_slice(w, DMLP, kb + 32, 0, bsm[buf ^ 1], tid);

        v16bf bf;
#pragma unroll
        for (int i = 0; i < 16; ++i)
            bf[i] = (__bf16)bsm[buf][kgrp * 8 + ((i < 8) ? i : (8 + i))][col];

#pragma unroll
        for (int mt = 0; mt < 5; ++mt) {
            const float* ap = outg + (size_t)(m0 + mt * 16 + arow) * DHID + kb + kgrp * 8;
            float af[16];
            *(float4*)(af + 0)  = *(const float4*)(ap);
            *(float4*)(af + 4)  = *(const float4*)(ap + 4);
            *(float4*)(af + 8)  = *(const float4*)(ap + 16);
            *(float4*)(af + 12) = *(const float4*)(ap + 20);
            const int k0 = kb + kgrp * 8;
            v16bf a;
#pragma unroll
            for (int i = 0; i < 16; ++i) {
                int k = k0 + ((i < 8) ? i : (16 + (i - 8)));
                float v = af[i] + gbias[k];
                v = (v > 0.f) ? v : (__expf(v) - 1.f);   // ELU
                a[i] = (__bf16)v;
            }
            acc[mt] = __builtin_amdgcn_wmma_f32_16x16x32_bf16(
                false, a, false, bf, (short)0, acc[mt], false, false);
        }
    }
#pragma unroll
    for (int mt = 0; mt < 5; ++mt)
#pragma unroll
        for (int r = 0; r < 8; ++r)
            hbuf[(size_t)(m0 + mt * 16 + r + kgrp * 8) * DMLP + col] = acc[mt][r] + mb[col];
}

// =====================================================================
// K7: BN0 column stats (biased var over N rows), one block per column
// =====================================================================
__global__ __launch_bounds__(256)
void bn_stats_kernel(const float* __restrict__ h,
                     float* __restrict__ meanv,
                     float* __restrict__ rstdv) {
    int c = blockIdx.x;                      // 0..127
    __shared__ float ssum[256], ssq[256];
    float s = 0.f, q = 0.f;
    for (int i = threadIdx.x; i < N_NODES; i += 256) {
        float v = h[(size_t)i * DMLP + c];
        s += v; q += v * v;
    }
    ssum[threadIdx.x] = s; ssq[threadIdx.x] = q;
    __syncthreads();
    for (int off = 128; off; off >>= 1) {
        if (threadIdx.x < off) {
            ssum[threadIdx.x] += ssum[threadIdx.x + off];
            ssq[threadIdx.x]  += ssq[threadIdx.x + off];
        }
        __syncthreads();
    }
    if (threadIdx.x == 0) {
        float m   = ssum[0] * (1.0f / N_NODES);
        float var = ssq[0] * (1.0f / N_NODES) - m * m;
        meanv[c] = m;
        rstdv[c] = rsqrtf(var + BN_EPS);
    }
}

// =====================================================================
// K8: apply BN0 and global_add_pool into hp[G,128]
// =====================================================================
__global__ void pool_kernel(const float* __restrict__ h,
                            const int* __restrict__ batch,
                            const float* __restrict__ meanv,
                            const float* __restrict__ rstdv,
                            const float* __restrict__ g,
                            const float* __restrict__ b,
                            float* __restrict__ hp) {
    int idx = blockIdx.x * blockDim.x + threadIdx.x;
    if (idx >= N_NODES * DMLP) return;
    int i = idx >> 7, c = idx & 127;
    float v = (h[idx] - meanv[c]) * rstdv[c] * g[c] + b[c];
    atomicAdd(&hp[batch[i] * DMLP + c], v);
}

// =====================================================================
// K9: small FC + row-BN (+optional ReLU). One block (64 threads) per col.
// =====================================================================
__global__ __launch_bounds__(64)
void fc_bn_kernel(const float* __restrict__ in,
                  const float* __restrict__ W,
                  const float* __restrict__ bias,
                  const float* __restrict__ gamma,
                  const float* __restrict__ beta,
                  float* __restrict__ out,
                  int K, int M, int do_relu) {
    int j = blockIdx.x;            // output column
    int g = threadIdx.x;           // graph row 0..63
    float acc = bias[j];
    const float* ir = in + (size_t)g * K;
    for (int k = 0; k < K; ++k) acc += ir[k] * W[(size_t)k * M + j];
    __shared__ float sv[NGRAPH];
    sv[g] = acc;
    __syncthreads();
    float m = 0.f, q = 0.f;
#pragma unroll 8
    for (int t = 0; t < NGRAPH; ++t) { m += sv[t]; q += sv[t] * sv[t]; }
    m *= (1.0f / NGRAPH);
    q = q * (1.0f / NGRAPH) - m * m;        // biased variance over rows
    float v = (acc - m) * rsqrtf(q + BN_EPS) * gamma[j] + beta[j];
    if (do_relu) v = fmaxf(v, 0.f);
    out[(size_t)g * M + j] = v;
}

// =====================================================================
// host driver
// =====================================================================
extern "C" void kernel_launch(void* const* d_in, const int* in_sizes, int n_in,
                              void* d_out, int out_size, void* d_ws, size_t ws_size,
                              hipStream_t stream) {
    const float* x       = (const float*)d_in[0];
    const float* gat_w   = (const float*)d_in[1];
    const float* a_src   = (const float*)d_in[2];
    const float* a_dst   = (const float*)d_in[3];
    const float* gat_b   = (const float*)d_in[4];
    const float* mlp_w   = (const float*)d_in[5];
    const float* mlp_b   = (const float*)d_in[6];
    const float* bn0_g   = (const float*)d_in[7];
    const float* bn0_b   = (const float*)d_in[8];
    const float* fc1_w   = (const float*)d_in[9];
    const float* fc1_b   = (const float*)d_in[10];
    const float* bn1_g   = (const float*)d_in[11];
    const float* bn1_b   = (const float*)d_in[12];
    const float* fc2_w   = (const float*)d_in[13];
    const float* fc2_b   = (const float*)d_in[14];
    const float* bn2_g   = (const float*)d_in[15];
    const float* bn2_b   = (const float*)d_in[16];
    const float* fc3_w   = (const float*)d_in[17];
    const float* fc3_b   = (const float*)d_in[18];
    const float* bn3_g   = (const float*)d_in[19];
    const float* bn3_b   = (const float*)d_in[20];
    const int*   ei      = (const int*)d_in[21];
    const int*   batch   = (const int*)d_in[22];

    // ---- workspace carve-up (~86 MB of fp32) ----
    float* ws     = (float*)d_ws;
    float* xp     = ws;                                           // N*256
    float* a_s    = xp    + (size_t)N_NODES * DHID;               // N*2
    float* a_d    = a_s   + (size_t)N_NODES * NHEAD;              // N*2
    unsigned* mkey= (unsigned*)(a_d + (size_t)N_NODES * NHEAD);   // N*2
    float* denom  = (float*)(mkey + (size_t)N_NODES * NHEAD);     // N*2
    float* ew     = denom + (size_t)N_NODES * NHEAD;              // (E+N)*2
    float* outg   = ew    + (size_t)(E_EDGES + N_NODES) * NHEAD;  // N*256
    float* hbuf   = outg  + (size_t)N_NODES * DHID;               // N*128
    float* meanv  = hbuf  + (size_t)N_NODES * DMLP;               // 128
    float* rstdv  = meanv + DMLP;                                 // 128
    float* hp     = rstdv + DMLP;                                 // 64*128
    float* z1     = hp    + (size_t)NGRAPH * DMLP;                // 64*512
    float* z2     = z1    + (size_t)NGRAPH * 512;                 // 64*256

    // K0: init accumulators (covers largest target: N*256)
    init_ws_kernel<<<(N_NODES * DHID + 255) / 256, 256, 0, stream>>>(
        mkey, denom, outg, hp);

    // K1: xp = x @ gat_w  (WMMA bf16, async-LDS double-buffered B)
    gemm_xp_kernel<<<dim3(N_NODES / BM, 2), 256, 0, stream>>>(x, gat_w, xp);

    // K2: attention logits
    attn_dots_kernel<<<(N_NODES * NHEAD + 255) / 256, 256, 0, stream>>>(
        xp, a_src, a_dst, a_s, a_d);

    // K3/K4: edge softmax (max, exp+sum)
    int nedge_th = (E_EDGES + N_NODES) * NHEAD;
    edge_max_kernel<<<(nedge_th + 255) / 256, 256, 0, stream>>>(ei, a_s, a_d, mkey);
    edge_exp_kernel<<<(nedge_th + 255) / 256, 256, 0, stream>>>(
        ei, a_s, a_d, mkey, ew, denom);

    // K5: aggregation (one wave per edge)
    int nwaves = E_EDGES + N_NODES;
    aggregate_kernel<<<(nwaves + 7) / 8, 256, 0, stream>>>(ei, xp, ew, denom, outg);

    // K6: h = elu(out + bias) @ mlp_w + mlp_b  (WMMA bf16, fused ELU)
    gemm_h_kernel<<<dim3(N_NODES / BM, 1), 256, 0, stream>>>(
        outg, gat_b, mlp_w, mlp_b, hbuf);

    // K7/K8: BN0 stats then pooled apply
    bn_stats_kernel<<<DMLP, 256, 0, stream>>>(hbuf, meanv, rstdv);
    pool_kernel<<<(N_NODES * DMLP + 255) / 256, 256, 0, stream>>>(
        hbuf, batch, meanv, rstdv, bn0_g, bn0_b, hp);

    // K9: FC stack on 64 graphs
    fc_bn_kernel<<<512, 64, 0, stream>>>(hp, fc1_w, fc1_b, bn1_g, bn1_b, z1, 128, 512, 1);
    fc_bn_kernel<<<256, 64, 0, stream>>>(z1, fc2_w, fc2_b, bn2_g, bn2_b, z2, 512, 256, 1);
    fc_bn_kernel<<<1,   64, 0, stream>>>(z2, fc3_w, fc3_b, bn3_g, bn3_b,
                                         (float*)d_out, 256, 1, 0);
    (void)in_sizes; (void)n_in; (void)out_size; (void)ws_size;
}
